// GameTheory_5025111736966
// MI455X (gfx1250) — matrix operand order repaired and checked by reading, not compiled, not run on hardware
//
#include <hip/hip_runtime.h>
#include <math.h>

typedef _Float16 half_t;
typedef __attribute__((ext_vector_type(16))) _Float16 v16h;
typedef __attribute__((ext_vector_type(8)))  float    v8f;

#define WMMA_F16(a, b, c) \
  __builtin_amdgcn_wmma_f32_16x16x32_f16(false, (a), false, (b), (short)0, (c), false, false)

// A-matrix (16-bit, 16x32) per-lane element -> K column. i = 2*vgpr + half.
__device__ __forceinline__ int a_col(int i, int g) {
  return i + (g << 4);
}

__device__ __forceinline__ unsigned int f2ord(float x) {
  unsigned int u = __float_as_uint(x);
  return (u & 0x80000000u) ? ~u : (u | 0x80000000u);
}

// ---------------------------------------------------------------------------
// One 16-row tile of the 3-layer MLP: X(16x64) -> H(16x32) -> relu G(16x32)
// -> S(16x16). One wave per tile; all matrix math through v_wmma.
// ---------------------------------------------------------------------------
__device__ __forceinline__ void mlp_tile(
    const float* __restrict__ emb, int row0,
    const float* __restrict__ wA, const float* __restrict__ bA,
    const float* __restrict__ w1, const float* __restrict__ b1,
    const float* __restrict__ w2, const float* __restrict__ b2,
    float* __restrict__ S, float* __restrict__ inv_norm,
    half_t* __restrict__ hbuf /* 16*32 per-wave LDS */) {
  const int lane = threadIdx.x & 31;
  const int g = lane >> 4;   // half-wave
  const int n = lane & 15;   // A-row m / B,C,D column

  // ---- layer 1: X(16x64) @ wA(64x32), K split into two 32-wide steps ----
  v8f acc0 = {}; v8f acc1 = {};
  const float* xrow = emb + (size_t)(row0 + n) * 64;
#pragma unroll
  for (int ks = 0; ks < 2; ++ks) {
    v16h a;
#pragma unroll
    for (int i = 0; i < 16; ++i)
      a[i] = (half_t)xrow[a_col(i, g) + 32 * ks];
    v16h bm0, bm1;
#pragma unroll
    for (int i = 0; i < 16; ++i) {
      int kr = i + 16 * g + 32 * ks;       // B: K = i + 16g, N = n (+16)
      bm0[i] = (half_t)wA[kr * 32 + n];
      bm1[i] = (half_t)wA[kr * 32 + n + 16];
    }
    acc0 = WMMA_F16(a, bm0, acc0);
    acc1 = WMMA_F16(a, bm1, acc1);
  }
  const float biasA0 = bA[n], biasA1 = bA[n + 16];
#pragma unroll
  for (int r = 0; r < 8; ++r) {            // C/D: M = r + 8g
    int M = r + 8 * g;
    hbuf[M * 32 + n]      = (half_t)(acc0[r] + biasA0);   // hm (no relu)
    hbuf[M * 32 + n + 16] = (half_t)(acc1[r] + biasA1);
  }

  // ---- layer 2: H(16x32) @ w1(32x32), +bias, relu ----
  v16h a2;
#pragma unroll
  for (int i = 0; i < 16; ++i)
    a2[i] = hbuf[n * 32 + a_col(i, g)];
  v16h bw0, bw1;
#pragma unroll
  for (int i = 0; i < 16; ++i) {
    int kr = i + 16 * g;
    bw0[i] = (half_t)w1[kr * 32 + n];
    bw1[i] = (half_t)w1[kr * 32 + n + 16];
  }
  v8f c0 = {}; v8f c1 = {};
  c0 = WMMA_F16(a2, bw0, c0);
  c1 = WMMA_F16(a2, bw1, c1);
  const float bias10 = b1[n], bias11 = b1[n + 16];
#pragma unroll
  for (int r = 0; r < 8; ++r) {
    int M = r + 8 * g;
    hbuf[M * 32 + n]      = (half_t)fmaxf(c0[r] + bias10, 0.0f);
    hbuf[M * 32 + n + 16] = (half_t)fmaxf(c1[r] + bias11, 0.0f);
  }

  // ---- layer 3: G(16x32) @ w2(32x16) ----
  v16h a3;
#pragma unroll
  for (int i = 0; i < 16; ++i)
    a3[i] = hbuf[n * 32 + a_col(i, g)];
  v16h bw2;
#pragma unroll
  for (int i = 0; i < 16; ++i)
    bw2[i] = (half_t)w2[(i + 16 * g) * 16 + n];
  v8f s = {};
  s = WMMA_F16(a3, bw2, s);

  const float bias2 = b2[n];
#pragma unroll
  for (int r = 0; r < 8; ++r) {
    float v = s[r] + bias2;
    int row = row0 + r + 8 * g;
    S[(size_t)row * 16 + n] = v;
    // row norm: reduce v^2 across the 16 lanes of this half-wave
    float sq = v * v;
#pragma unroll
    for (int off = 1; off < 16; off <<= 1)
      sq += __shfl_xor(sq, off, 32);
    if (n == 0)
      inv_norm[row] = rsqrtf(sq);
  }
}

__global__ void mlp_tables_kernel(
    const float* __restrict__ embM, const float* __restrict__ embD,
    const float* __restrict__ w_m, const float* __restrict__ b_m,
    const float* __restrict__ w_d, const float* __restrict__ b_d,
    const float* __restrict__ msw1, const float* __restrict__ msb1,
    const float* __restrict__ msw2, const float* __restrict__ msb2,
    const float* __restrict__ dsw1, const float* __restrict__ dsb1,
    const float* __restrict__ dsw2, const float* __restrict__ dsb2,
    float* __restrict__ S_M, float* __restrict__ inv_nm,
    float* __restrict__ S_D, float* __restrict__ inv_nd,
    int tiles_m, int tiles_total) {
  __shared__ half_t lds[8][16 * 32];
  const int wave = threadIdx.x >> 5;
  const int tile = blockIdx.x * 8 + wave;     // wave-uniform: EXEC stays all-1s
  if (tile >= tiles_total) return;
  if (tile < tiles_m)
    mlp_tile(embM, tile * 16, w_m, b_m, msw1, msb1, msw2, msb2,
             S_M, inv_nm, lds[wave]);
  else
    mlp_tile(embD, (tile - tiles_m) * 16, w_d, b_d, dsw1, dsb1, dsw2, dsb2,
             S_D, inv_nd, lds[wave]);
}

// ---------------------------------------------------------------------------
__global__ void zero_kernel(unsigned int* __restrict__ p, int nwords) {
  int i = blockIdx.x * blockDim.x + threadIdx.x;
  if (i < nwords) p[i] = 0u;
}

// ---------------------------------------------------------------------------
// payoff + scatter keys + bitmap + BCE partials
// ---------------------------------------------------------------------------
__global__ void payoff_kernel(
    const int* __restrict__ miI, const int* __restrict__ diI,
    const float* __restrict__ labels,
    const float* __restrict__ S_M, const float* __restrict__ inv_nm,
    const float* __restrict__ S_D, const float* __restrict__ inv_nd,
    float* __restrict__ out_payoff,
    unsigned long long* __restrict__ rowmax,
    unsigned int* __restrict__ bitmap, int words_per_row,
    double* __restrict__ bce_part, int N) {
  const int i = blockIdx.x * blockDim.x + threadIdx.x;
  double bce = 0.0;
  if (i < N) {
    const int mi = miI[i], di = diI[i];
    const float4* sm = (const float4*)(S_M + (size_t)mi * 16);
    const float4* sd = (const float4*)(S_D + (size_t)di * 16);
    float dot = 0.0f;
#pragma unroll
    for (int k = 0; k < 4; ++k) {
      float4 a = sm[k], b = sd[k];
      dot += a.x * b.x + a.y * b.y + a.z * b.z + a.w * b.w;
    }
    const float x = dot * inv_nm[mi] * inv_nd[di];
    out_payoff[i] = x;
    // packed key: max payoff wins, smallest column wins ties (JAX argmax-first)
    unsigned long long key =
        ((unsigned long long)f2ord(x) << 32) | (0xFFFFFFFFu - (unsigned int)di);
    atomicMax(&rowmax[mi], key);
    atomicOr(&bitmap[(size_t)mi * words_per_row + (di >> 5)], 1u << (di & 31));
    const float y = labels[i];
    bce = (double)(fmaxf(x, 0.0f) - x * y + log1pf(expf(-fabsf(x))));
  }
  __shared__ double red[256];
  red[threadIdx.x] = bce;
  __syncthreads();
  for (int s = 128; s > 0; s >>= 1) {
    if (threadIdx.x < s) red[threadIdx.x] += red[threadIdx.x + s];
    __syncthreads();
  }
  if (threadIdx.x == 0) bce_part[blockIdx.x] = red[0];
}

// ---------------------------------------------------------------------------
// per-row argmax: positive max -> its column; else first unset (zero) cell
// ---------------------------------------------------------------------------
__global__ void argmax_kernel(const unsigned long long* __restrict__ rowmax,
                              const unsigned int* __restrict__ bitmap,
                              int* __restrict__ best_idx,
                              int num_m, int words_per_row) {
  const int m = blockIdx.x * blockDim.x + threadIdx.x;
  if (m >= num_m) return;
  const unsigned long long key = rowmax[m];
  const unsigned int ord = (unsigned int)(key >> 32);
  int best = 0;
  if (ord > 0x80000000u) {  // max payoff > 0.0
    best = (int)(0xFFFFFFFFu - (unsigned int)(key & 0xFFFFFFFFull));
  } else {                  // row max is an unset 0.0 cell: first clear bit
    const unsigned int* bm = bitmap + (size_t)m * words_per_row;
    for (int w = 0; w < words_per_row; ++w) {
      unsigned int word = bm[w];
      if (word != 0xFFFFFFFFu) { best = w * 32 + __builtin_ctz(~word); break; }
    }
  }
  best_idx[m] = best;
}

// ---------------------------------------------------------------------------
// nash sums of squared diffs vs the "best" strategy rows
// ---------------------------------------------------------------------------
__global__ void nash_kernel(
    const int* __restrict__ miI, const int* __restrict__ diI,
    const int* __restrict__ best_idx,
    const float* __restrict__ S_M, const float* __restrict__ S_D,
    double* __restrict__ mss_part, double* __restrict__ dss_part, int N) {
  const int i = blockIdx.x * blockDim.x + threadIdx.x;
  double am = 0.0, ad = 0.0;
  if (i < N) {
    const int mi = miI[i], di = diI[i];
    int p = best_idx[mi];
    if (p > N - 1) p = N - 1;   // JAX gather clamps OOB
    const int mp = miI[p], dp = diI[p];
    const float* a = S_M + (size_t)mi * 16;
    const float* b = S_M + (size_t)mp * 16;
    const float* c = S_D + (size_t)di * 16;
    const float* d = S_D + (size_t)dp * 16;
    float sm = 0.0f, sd = 0.0f;
#pragma unroll
    for (int k = 0; k < 16; ++k) {
      float e = a[k] - b[k]; sm += e * e;
      float f = c[k] - d[k]; sd += f * f;
    }
    am = (double)sm; ad = (double)sd;
  }
  __shared__ double red[256];
  red[threadIdx.x] = am;
  __syncthreads();
  for (int s = 128; s > 0; s >>= 1) {
    if (threadIdx.x < s) red[threadIdx.x] += red[threadIdx.x + s];
    __syncthreads();
  }
  if (threadIdx.x == 0) mss_part[blockIdx.x] = red[0];
  __syncthreads();
  red[threadIdx.x] = ad;
  __syncthreads();
  for (int s = 128; s > 0; s >>= 1) {
    if (threadIdx.x < s) red[threadIdx.x] += red[threadIdx.x + s];
    __syncthreads();
  }
  if (threadIdx.x == 0) dss_part[blockIdx.x] = red[0];
}

// ---------------------------------------------------------------------------
// fixed-order final reduction (deterministic) + loss
// ---------------------------------------------------------------------------
__global__ void finalize_kernel(const double* __restrict__ bce_part,
                                const double* __restrict__ mss_part,
                                const double* __restrict__ dss_part,
                                int nparts, float* __restrict__ out_loss, int N) {
  __shared__ double red[256];
  double b = 0.0, m = 0.0, d = 0.0;
  for (int j = threadIdx.x; j < nparts; j += 256) {
    b += bce_part[j]; m += mss_part[j]; d += dss_part[j];
  }
  double sums[3] = {b, m, d}, tot[3];
  for (int q = 0; q < 3; ++q) {
    red[threadIdx.x] = sums[q];
    __syncthreads();
    for (int s = 128; s > 0; s >>= 1) {
      if (threadIdx.x < s) red[threadIdx.x] += red[threadIdx.x + s];
      __syncthreads();
    }
    tot[q] = red[0];
    __syncthreads();
  }
  if (threadIdx.x == 0) {
    const double dn = (double)N;
    double nash = 0.5 * ((tot[1] + tot[2]) / (16.0 * dn));
    out_loss[0] = (float)(nash + tot[0] / dn);
  }
}

// ---------------------------------------------------------------------------
extern "C" void kernel_launch(void* const* d_in, const int* in_sizes, int n_in,
                              void* d_out, int out_size, void* d_ws, size_t ws_size,
                              hipStream_t stream) {
  (void)n_in; (void)out_size; (void)ws_size;
  const float* embM = (const float*)d_in[0];
  const float* embD = (const float*)d_in[1];
  const int*   miI  = (const int*)d_in[2];
  const int*   diI  = (const int*)d_in[3];
  const float* lab  = (const float*)d_in[4];
  const float* w_m  = (const float*)d_in[5];  const float* b_m  = (const float*)d_in[6];
  const float* w_d  = (const float*)d_in[7];  const float* b_d  = (const float*)d_in[8];
  const float* msw1 = (const float*)d_in[9];  const float* msb1 = (const float*)d_in[10];
  const float* msw2 = (const float*)d_in[11]; const float* msb2 = (const float*)d_in[12];
  const float* dsw1 = (const float*)d_in[13]; const float* dsb1 = (const float*)d_in[14];
  const float* dsw2 = (const float*)d_in[15]; const float* dsb2 = (const float*)d_in[16];

  const int num_m = in_sizes[0] / 64;   // 2048
  const int num_d = in_sizes[1] / 64;   // 4096
  const int N     = in_sizes[2];        // 1,000,000
  const int NB    = (N + 255) / 256;
  const int wpr   = (num_d + 31) / 32;  // bitmap words per row

  char* ws = (char*)d_ws;
  size_t off = 0;
  auto take = [&](size_t bytes) -> void* {
    void* p = ws + off;
    off = (off + bytes + 255) & ~(size_t)255;
    return p;
  };
  float* S_M = (float*)take((size_t)num_m * 16 * 4);
  float* S_D = (float*)take((size_t)num_d * 16 * 4);
  float* inv_nm = (float*)take((size_t)num_m * 4);
  float* inv_nd = (float*)take((size_t)num_d * 4);
  unsigned long long* rowmax = (unsigned long long*)take((size_t)num_m * 8);
  unsigned int* bitmap = (unsigned int*)take((size_t)num_m * wpr * 4);
  int* best = (int*)take((size_t)num_m * 4);
  double* bceP = (double*)take((size_t)NB * 8);
  double* mssP = (double*)take((size_t)NB * 8);
  double* dssP = (double*)take((size_t)NB * 8);

  float* out = (float*)d_out;           // payoff [N] then loss scalar

  // clear scatter state (harness does not re-poison between replays)
  {
    int w1 = num_m * 2;                 // rowmax as u32 words
    zero_kernel<<<(w1 + 255) / 256, 256, 0, stream>>>((unsigned int*)rowmax, w1);
    int w2 = num_m * wpr;
    zero_kernel<<<(w2 + 255) / 256, 256, 0, stream>>>(bitmap, w2);
  }

  // WMMA MLP tables
  const int tiles_m = num_m / 16;
  const int tiles_total = tiles_m + num_d / 16;
  mlp_tables_kernel<<<(tiles_total + 7) / 8, 256, 0, stream>>>(
      embM, embD, w_m, b_m, w_d, b_d, msw1, msb1, msw2, msb2,
      dsw1, dsb1, dsw2, dsb2, S_M, inv_nm, S_D, inv_nd, tiles_m, tiles_total);

  payoff_kernel<<<NB, 256, 0, stream>>>(miI, diI, lab, S_M, inv_nm, S_D, inv_nd,
                                        out, rowmax, bitmap, wpr, bceP, N);

  argmax_kernel<<<(num_m + 255) / 256, 256, 0, stream>>>(rowmax, bitmap, best,
                                                         num_m, wpr);

  nash_kernel<<<NB, 256, 0, stream>>>(miI, diI, best, S_M, S_D, mssP, dssP, N);

  finalize_kernel<<<1, 256, 0, stream>>>(bceP, mssP, dssP, NB, out + N, N);
}
